// VQVAE_26903675142238
// MI455X (gfx1250) — matrix operand order repaired and checked by reading, not compiled, not run on hardware
//
#include <hip/hip_runtime.h>
#include <hip/hip_fp16.h>

typedef __attribute__((ext_vector_type(16))) _Float16 v16h;
typedef __attribute__((ext_vector_type(8)))  _Float16 v8h;
typedef __attribute__((ext_vector_type(8)))  float    v8f;
typedef __attribute__((ext_vector_type(4)))  int      v4i;

union HV { v16h v; v8h h[2]; };

#define WMMA(a,b,c) __builtin_amdgcn_wmma_f32_16x16x32_f16(false,(a),false,(b),(short)0,(c),false,false)

#if defined(__gfx1250__) && __has_builtin(__builtin_amdgcn_global_load_async_to_lds_b128)
#define HAS_ASYNC_LDS 1
#else
#define HAS_ASYNC_LDS 0
#endif

#if HAS_ASYNC_LDS
typedef v4i __attribute__((address_space(1))) *as1_v4i;   // global (prints as __device__)
typedef v4i __attribute__((address_space(3))) *as3_v4i;   // LDS    (prints as __shared__)
#endif

// ---- async wait helpers: builtin if declared, else exact ISA mnemonic ----
__device__ __forceinline__ void wait_async0() {
#if HAS_ASYNC_LDS
#if __has_builtin(__builtin_amdgcn_s_wait_asynccnt)
  __builtin_amdgcn_s_wait_asynccnt(0);
#else
  asm volatile("s_wait_asynccnt 0x0" ::: "memory");
#endif
#endif
}
__device__ __forceinline__ void wait_async2() {
#if HAS_ASYNC_LDS
#if __has_builtin(__builtin_amdgcn_s_wait_asynccnt)
  __builtin_amdgcn_s_wait_asynccnt(2);
#else
  asm volatile("s_wait_asynccnt 0x2" ::: "memory");
#endif
#endif
}

// 16B global -> LDS copy (async if available)
__device__ __forceinline__ void cp16(const _Float16* g, _Float16* l) {
#if HAS_ASYNC_LDS
  __builtin_amdgcn_global_load_async_to_lds_b128((as1_v4i)g, (as3_v4i)l, 0, 0);
#else
  *(v8h*)l = *(const v8h*)g;
#endif
}

// A-fragment loader: 16-bit A 16x32 layout. Lane L: m = L&15, chunk offset 16*(L>>4).
__device__ __forceinline__ v16h ldA(const _Float16* __restrict__ p, int kk) {
  HV u;
  u.h[0] = *(const v8h*)(p + kk);
  u.h[1] = *(const v8h*)(p + kk + 8);
  return u.v;
}
// B-fragment loader from LDS (16B-aligned pair)
__device__ __forceinline__ v16h ldB16(const _Float16* p) {
  HV u;
  u.h[0] = *(const v8h*)(p);
  u.h[1] = *(const v8h*)(p + 8);
  return u.v;
}

#define BROW 40   // padded LDS row stride in halves (80B): conflict-free frag reads

// ---------------------------------------------------------------------------
// GEMM: C[M,N] = A[M,K] (f16 row major) x Bt[N,K] (f16 row major = B^T)
// Block = 128 thr = 4 waves. Wave tile 32x64 (2 A frags x 4 N tiles).
// Block tile 128x64. B tile (64x32 halves) staged in LDS, double-buffered
// via async global->LDS copies.
// EPI 0: f32 + f16 store, no bias | EPI 1: relu+bias -> f16 | EPI 2: sigmoid -> f32
// ---------------------------------------------------------------------------
template<int EPI>
__global__ __launch_bounds__(128, 2)
void gemm16(const _Float16* __restrict__ A, const _Float16* __restrict__ Bt,
            const float* __restrict__ bias, int bshift, int N, int K,
            _Float16* __restrict__ outH, float* __restrict__ outF)
{
  __shared__ _Float16 Bs[2][64 * BROW];

  const int lane  = threadIdx.x & 31;
  const int wave  = threadIdx.x >> 5;
  const int lm    = lane & 15;
  const int hi    = lane >> 4;
  const int m0    = blockIdx.x * 128 + wave * 32;
  const int nbase = blockIdx.y * 64;

  const _Float16* Ap0 = A + (size_t)(m0 + lm) * K + 16 * hi;
  const _Float16* Ap1 = Ap0 + (size_t)16 * K;

  // staging map: thread t covers chunks t and t+128 (row = c>>2, 16B col = c&3)
  const int t  = threadIdx.x;
  const int sr = t >> 2;
  const int sq = t & 3;
  const _Float16* g0 = Bt + (size_t)(nbase + sr) * K + sq * 8;
  const _Float16* g1 = Bt + (size_t)(nbase + sr + 32) * K + sq * 8;
  _Float16* l0[2] = { &Bs[0][sr * BROW + sq * 8],        &Bs[1][sr * BROW + sq * 8] };
  _Float16* l1[2] = { &Bs[0][(sr + 32) * BROW + sq * 8], &Bs[1][(sr + 32) * BROW + sq * 8] };

  v8f c0[4] = {}, c1[4] = {};

  cp16(g0, l0[0]);
  cp16(g1, l1[0]);
  int cur = 0;
  for (int kk = 0; kk < K; kk += 32) {
    const bool nx = (kk + 32 < K);
    if (nx) { cp16(g0 + kk + 32, l0[cur ^ 1]); cp16(g1 + kk + 32, l1[cur ^ 1]); }
    if (nx) wait_async2(); else wait_async0();
    __syncthreads();                    // B tile [cur] visible to all waves

    __builtin_prefetch(Ap0 + kk + 256, 0, 0);   // global_prefetch_b8
    v16h a0 = ldA(Ap0, kk);
    v16h a1 = ldA(Ap1, kk);
    const _Float16* bp = &Bs[cur][lm * BROW + hi * 16];
#pragma unroll
    for (int j = 0; j < 4; ++j) {
      v16h b = ldB16(bp + j * 16 * BROW);
      c0[j] = WMMA(a0, b, c0[j]);
      c1[j] = WMMA(a1, b, c1[j]);
    }
    __syncthreads();                    // done reading [cur] before restage
    cur ^= 1;
  }

#pragma unroll
  for (int j = 0; j < 4; ++j) {
    const int n = nbase + 16 * j + lm;
    const float bb = (EPI != 0) ? bias[n >> bshift] : 0.0f;
#pragma unroll
    for (int v = 0; v < 8; ++v) {
      const int ma = m0 + v + 8 * hi;
      const int mb = ma + 16;
      float ra = c0[j][v];
      float rb = c1[j][v];
      if (EPI == 0) {
        outF[(size_t)ma * N + n] = ra;  outH[(size_t)ma * N + n] = (_Float16)ra;
        outF[(size_t)mb * N + n] = rb;  outH[(size_t)mb * N + n] = (_Float16)rb;
      } else if (EPI == 1) {
        ra += bb; ra = ra > 0.0f ? ra : 0.0f;
        rb += bb; rb = rb > 0.0f ? rb : 0.0f;
        outH[(size_t)ma * N + n] = (_Float16)ra;
        outH[(size_t)mb * N + n] = (_Float16)rb;
      } else {
        ra += bb; rb += bb;
        outF[(size_t)ma * N + n] = 1.0f / (1.0f + __expf(-ra));
        outF[(size_t)mb * N + n] = 1.0f / (1.0f + __expf(-rb));
      }
    }
  }
}

// ---------------------------------------------------------------------------
// VQ nearest-code: argmin_n (|c_n|^2 - 2 z.c_n).  K=64, N=512.
// ---------------------------------------------------------------------------
__global__ __launch_bounds__(128, 2)
void vq_argmin(const _Float16* __restrict__ zh, const _Float16* __restrict__ cbh,
               const float* __restrict__ cnorm, int* __restrict__ idxI,
               float* __restrict__ idxF)
{
  const int lane  = threadIdx.x & 31;
  const int wave  = threadIdx.x >> 5;
  const int lm    = lane & 15;
  const int hi    = lane >> 4;
  const int mbase = blockIdx.x * 64 + wave * 16;

  const _Float16* Ap = zh + (size_t)(mbase + lm) * 64 + 16 * hi;
  const v16h a0 = ldA(Ap, 0);
  const v16h a1 = ldA(Ap, 32);

  float bv[8]; int bi[8];
#pragma unroll
  for (int v = 0; v < 8; ++v) { bv[v] = 3.4e38f; bi[v] = 0x7fffffff; }

  for (int ti = 0; ti < 32; ++ti) {
    const int n = ti * 16 + lm;
    const _Float16* Bp = cbh + (size_t)n * 64 + 16 * hi;
    v16h b0 = *(const v16h*)(Bp);
    v16h b1 = *(const v16h*)(Bp + 32);
    v8f c = {};
    c = WMMA(a0, b0, c);
    c = WMMA(a1, b1, c);
    const float cn = cnorm[n];
#pragma unroll
    for (int v = 0; v < 8; ++v) {
      float d = cn - 2.0f * c[v];
      if (d < bv[v] || (d == bv[v] && n < bi[v])) { bv[v] = d; bi[v] = n; }
    }
  }

#pragma unroll
  for (int v = 0; v < 8; ++v) {
    float d = bv[v]; int n = bi[v];
    for (int mask = 1; mask < 16; mask <<= 1) {
      float od = __shfl_xor(d, mask, 32);
      int   on = __shfl_xor(n, mask, 32);
      if (od < d || (od == d && on < n)) { d = od; n = on; }
    }
    if (lm == 0) {
      const int m = mbase + v + 8 * hi;
      idxI[m] = n;
      idxF[m] = (float)n;
    }
  }
}

// ---------------------------------------------------------------------------
// Prep kernels
// ---------------------------------------------------------------------------
__global__ void cast_f16(const float* __restrict__ in, _Float16* __restrict__ out, size_t n) {
  size_t i = (size_t)blockIdx.x * blockDim.x + threadIdx.x;
  if (i < n) out[i] = (_Float16)in[i];
}

__global__ void trans_wenc(const float* __restrict__ in, _Float16* __restrict__ out) {
  int i = blockIdx.x * blockDim.x + threadIdx.x;       // 64*768
  if (i >= 64 * 768) return;
  int n = i / 768, k = i % 768;
  out[i] = (_Float16)in[k * 64 + n];
}

__global__ void row_sumsq(const float* __restrict__ cb, float* __restrict__ cn) {
  int r = blockIdx.x * blockDim.x + threadIdx.x;
  if (r < 512) {
    float s = 0.f;
    for (int j = 0; j < 64; ++j) { float v = cb[r * 64 + j]; s += v * v; }
    cn[r] = s;
  }
}

// Transposed deconv weight Wt[N,K]: N = OC*outS^2, K = IC*inS^2.
__global__ void build_w(const float* __restrict__ w, _Float16* __restrict__ wt,
                        int OC, int IC, int inS, int outS)
{
  const size_t K = (size_t)IC * inS * inS;
  const size_t N = (size_t)OC * outS * outS;
  size_t i = (size_t)blockIdx.x * blockDim.x + threadIdx.x;
  if (i >= N * K) return;
  int k = (int)(i % K), n = (int)(i / K);
  int oc = n / (outS * outS), oy = (n / outS) % outS, ox = n % outS;
  int ic = k / (inS * inS),   iy = (k / inS) % inS,   ix = k % inS;
  int ky = 2 + 2 * iy - oy, kx = 2 + 2 * ix - ox;
  float v = 0.f;
  if (ky >= 0 && ky < 4 && kx >= 0 && kx < 4)
    v = w[(((size_t)oc * IC + ic) * 4 + (3 - ky)) * 4 + (3 - kx)];
  wt[i] = (_Float16)v;
}

__global__ void vq_gather(const int* __restrict__ idx, const float* __restrict__ cb,
                          float* __restrict__ zqF, _Float16* __restrict__ zqH, int n)
{
  int i = blockIdx.x * blockDim.x + threadIdx.x;
  if (i >= n) return;
  int b = i >> 6, d = i & 63;
  float v = cb[(size_t)idx[b] * 64 + d];
  zqF[i] = v;
  zqH[i] = (_Float16)v;
}

// ---------------------------------------------------------------------------
extern "C" void kernel_launch(void* const* d_in, const int* in_sizes, int n_in,
                              void* d_out, int out_size, void* d_ws, size_t ws_size,
                              hipStream_t stream)
{
  (void)in_sizes; (void)n_in; (void)out_size;
  const float* x    = (const float*)d_in[0];
  const float* Wenc = (const float*)d_in[1];
  const float* cbk  = (const float*)d_in[2];
  const float* w1 = (const float*)d_in[3];  const float* b1 = (const float*)d_in[4];
  const float* w2 = (const float*)d_in[5];  const float* b2 = (const float*)d_in[6];
  const float* w3 = (const float*)d_in[7];  const float* b3 = (const float*)d_in[8];
  const float* w4 = (const float*)d_in[9];  const float* b4 = (const float*)d_in[10];

  const int B = 16384;
  float* out    = (float*)d_out;
  float* xrec   = out;                          // B*768
  float* zout   = out + (size_t)B * 768;        // B*64
  float* zqout  = zout + (size_t)B * 64;        // B*64
  float* idxout = zqout + (size_t)B * 64;       // B

  char* wsb = (char*)d_ws;
  size_t off = 0;
  auto take = [&](size_t bytes) -> char* {
    char* p = wsb + off;
    off += (bytes + 255) & ~(size_t)255;
    return p;
  };

  // -------- fixed (weight) scratch --------
  _Float16* WencT = (_Float16*)take((size_t)64 * 768 * 2);
  _Float16* cbH   = (_Float16*)take((size_t)512 * 64 * 2);
  float*    cnorm = (float*)   take(512 * 4);
  _Float16* W1t   = (_Float16*)take((size_t)1024 * 64 * 2);
  _Float16* W2t   = (_Float16*)take((size_t)2048 * 1024 * 2);
  _Float16* W3t   = (_Float16*)take((size_t)4096 * 2048 * 2);
  _Float16* W4t   = (_Float16*)take((size_t)768 * 4096 * 2);
  const size_t fixedBytes = off;

  // -------- adaptive batch chunk (multiple of 128) --------
  const size_t perRow = 2 * (768 + 64 + 64 + 1024 + 2048 + 4096) + 4 + 64;
  int CH = 2048;
  if (ws_size > fixedBytes + 8192) {
    size_t rows = (ws_size - fixedBytes - 8192) / perRow;
    if (rows < (size_t)CH) CH = (int)(rows & ~(size_t)127);
  } else CH = 128;
  if (CH < 128) CH = 128;

  _Float16* Xh  = (_Float16*)take((size_t)CH * 768 * 2);
  _Float16* zh  = (_Float16*)take((size_t)CH * 64 * 2);
  _Float16* zqh = (_Float16*)take((size_t)CH * 64 * 2);
  _Float16* h1  = (_Float16*)take((size_t)CH * 1024 * 2);
  _Float16* h2  = (_Float16*)take((size_t)CH * 2048 * 2);
  _Float16* h3  = (_Float16*)take((size_t)CH * 4096 * 2);
  int*      idxI = (int*)    take((size_t)CH * 4);

  // -------- one-time weight transforms --------
  cast_f16 <<<(512 * 64 + 255) / 256, 256, 0, stream>>>(cbk, cbH, (size_t)512 * 64);
  trans_wenc<<<(64 * 768 + 255) / 256, 256, 0, stream>>>(Wenc, WencT);
  row_sumsq<<<2, 256, 0, stream>>>(cbk, cnorm);
  build_w<<<(unsigned)(((size_t)1024 * 64   + 255) / 256), 256, 0, stream>>>(w1, W1t, 256, 64,  1, 2);
  build_w<<<(unsigned)(((size_t)2048 * 1024 + 255) / 256), 256, 0, stream>>>(w2, W2t, 128, 256, 2, 4);
  build_w<<<(unsigned)(((size_t)4096 * 2048 + 255) / 256), 256, 0, stream>>>(w3, W3t, 64,  128, 4, 8);
  build_w<<<(unsigned)(((size_t)768 * 4096  + 255) / 256), 256, 0, stream>>>(w4, W4t, 3,   64,  8, 16);

  // -------- pipeline over batch chunks --------
  const dim3 blk(128);
  for (int base = 0; base < B; base += CH) {
    const int M = (B - base < CH) ? (B - base) : CH;
    cast_f16<<<(unsigned)(((size_t)M * 768 + 255) / 256), 256, 0, stream>>>(
        x + (size_t)base * 768, Xh, (size_t)M * 768);

    // encoder: z = Xh @ WencT^T   (M x 768 x 64)
    gemm16<0><<<dim3(M / 128, 1), blk, 0, stream>>>(
        Xh, WencT, nullptr, 0, 64, 768, zh, zout + (size_t)base * 64);

    // VQ search + gather
    vq_argmin<<<dim3(M / 64), blk, 0, stream>>>(zh, cbH, cnorm, idxI, idxout + base);
    vq_gather<<<(unsigned)(((size_t)M * 64 + 255) / 256), 256, 0, stream>>>(
        idxI, cbk, zqout + (size_t)base * 64, zqh, M * 64);

    // decoder deconvs as GEMMs
    gemm16<1><<<dim3(M / 128, 1024 / 64), blk, 0, stream>>>(zqh, W1t, b1, 2, 1024, 64,   h1, nullptr);
    gemm16<1><<<dim3(M / 128, 2048 / 64), blk, 0, stream>>>(h1,  W2t, b2, 4, 2048, 1024, h2, nullptr);
    gemm16<1><<<dim3(M / 128, 4096 / 64), blk, 0, stream>>>(h2,  W3t, b3, 6, 4096, 2048, h3, nullptr);
    gemm16<2><<<dim3(M / 128, 768 / 64),  blk, 0, stream>>>(h3,  W4t, b4, 8, 768,  4096, nullptr,
                                                            xrec + (size_t)base * 768);
  }
}
